// QuantizedGRU_73684458930261
// MI455X (gfx1250) — matrix operand order repaired and checked by reading, not compiled, not run on hardware
//
#include <hip/hip_runtime.h>

// ---------------------------------------------------------------------------
// QuantizedGRU for MI455X (gfx1250): fp16-quantization == fp16 cast, so the
// whole GRU runs on V_WMMA_F32_16X16X32_F16. Reset gate is dead in the
// reference -> only z/n slices (rows 512..1535) of each weight are used.
// Persistent kernel, 16 WGs x 8 waves, grid barrier between phases.
// ---------------------------------------------------------------------------

typedef __attribute__((ext_vector_type(16))) _Float16 v16h;
typedef __attribute__((ext_vector_type(8)))  _Float16 v8h;
typedef __attribute__((ext_vector_type(4)))  _Float16 v4h;
typedef __attribute__((ext_vector_type(8)))  float    v8f;
typedef __attribute__((ext_vector_type(4)))  float    v4f;

constexpr int Bsz = 64, Ssz = 1024, Isz = 256, Hsz = 512;
constexpr int K0   = Isz + Hsz;   // 768  : [x_t | h0]
constexpr int K1   = 2 * Hsz;     // 1024 : [h0 | h1]
constexpr int LDA0 = K0 + 8;      // half-stride 776 -> word-stride 388 == 4 mod 64 (no bank conflicts)
constexpr int LDA1 = K1 + 8;      // half-stride 1032 -> word-stride 516 == 4 mod 64
constexpr int NWG  = 16;          // persistent workgroups (128 waves = 4 Mtiles x 32 Jtiles)
constexpr int TPB  = 256;         // 8 wave32 per workgroup

__device__ __forceinline__ v16h cat8(v8h lo, v8h hi) {
  return __builtin_shufflevector(lo, hi, 0,1,2,3,4,5,6,7,8,9,10,11,12,13,14,15);
}
__device__ __forceinline__ float q16(float x) { return (float)(_Float16)x; }

// ---------------- quantize x -> fp16 (RTE, exactly the reference format) ----
__global__ __launch_bounds__(256) void k_quant_x(const float* __restrict__ x,
                                                 _Float16* __restrict__ qx, int n) {
  int i = (blockIdx.x * 256 + threadIdx.x) * 4;
  if (i + 3 < n) {
    v4f v = *(const v4f*)(x + i);
    v4h h;
    h.x = (_Float16)v.x; h.y = (_Float16)v.y; h.z = (_Float16)v.z; h.w = (_Float16)v.w;
    *(v4h*)(qx + i) = h;
  }
}

// ---------------- pack z/n-only fp16 weights + fused quantized biases -------
// qW0 [1024 x 768]  : row r<512 -> (z) src row 512+r ; r>=512 -> (n) src row 1024+(r-512)
//                     cols [0,256) from w_ih_0, [256,768) from w_hh_0
// qW1 [1024 x 1024] : cols [0,512) from w_ih_1 (acts on h0), [512,1024) from w_hh_1
__global__ __launch_bounds__(256) void k_pack(
    const float* __restrict__ wih0, const float* __restrict__ whh0,
    const float* __restrict__ bih0, const float* __restrict__ bhh0,
    const float* __restrict__ wih1, const float* __restrict__ whh1,
    const float* __restrict__ bih1, const float* __restrict__ bhh1,
    _Float16* __restrict__ qW0, _Float16* __restrict__ qW1,
    float* __restrict__ b0, float* __restrict__ b1) {
  const int n0 = 1024 * K0, n1 = 1024 * K1;
  const int tot = n0 + n1 + 1024 + 1024;
  for (int i = blockIdx.x * 256 + threadIdx.x; i < tot; i += gridDim.x * 256) {
    if (i < n0) {
      int r = i / K0, c = i % K0;
      int sr = (r < Hsz) ? (Hsz + r) : (2 * Hsz + (r - Hsz));
      float v = (c < Isz) ? wih0[sr * Isz + c] : whh0[sr * Hsz + (c - Isz)];
      qW0[i] = (_Float16)v;
    } else if (i < n0 + n1) {
      int k = i - n0, r = k / K1, c = k % K1;
      int sr = (r < Hsz) ? (Hsz + r) : (2 * Hsz + (r - Hsz));
      float v = (c < Hsz) ? wih1[sr * Hsz + c] : whh1[sr * Hsz + (c - Hsz)];
      qW1[k] = (_Float16)v;
    } else if (i < n0 + n1 + 1024) {
      int r = i - n0 - n1;
      int sr = (r < Hsz) ? (Hsz + r) : (2 * Hsz + (r - Hsz));
      b0[r] = q16(bih0[sr]) + q16(bhh0[sr]);
    } else {
      int r = i - n0 - n1 - 1024;
      int sr = (r < Hsz) ? (Hsz + r) : (2 * Hsz + (r - Hsz));
      b1[r] = q16(bih1[sr]) + q16(bhh1[sr]);
    }
  }
}

// ---------------- zero hidden state (both parities) + barrier ---------------
__global__ __launch_bounds__(256) void k_init(float* h0f, float* h1f,
                                              _Float16* h0h, _Float16* h1h,
                                              unsigned* bar) {
  int i = blockIdx.x * 256 + threadIdx.x;
  const int nf = 2 * Bsz * Hsz;
  if (i < nf) {
    h0f[i] = 0.f; h1f[i] = 0.f;
    h0h[i] = (_Float16)0.f; h1h[i] = (_Float16)0.f;
  }
  if (i < 2) bar[i] = 0u;
}

// ---------------- grid-wide sense barrier (cluster barrier NOPs without a
// cluster dispatch, so real sync uses device-scope atomics) ------------------
__device__ __forceinline__ void grid_sync(unsigned* cnt, volatile unsigned* gen) {
  __syncthreads();
  if (threadIdx.x == 0) {
    __threadfence();                       // release phase writes to device scope
    unsigned g = *gen;
    if (atomicAdd(cnt, 1u) == (unsigned)(NWG - 1)) {
      *cnt = 0u;
      __threadfence();
      *gen = g + 1u;                       // publish
    } else {
      while (*gen == g) __builtin_amdgcn_s_sleep(1);
      __threadfence();                     // acquire
    }
  }
  __syncthreads();
}

// ---------------- persistent GRU kernel -------------------------------------
__global__ void __launch_bounds__(TPB, 1)
k_gru(const _Float16* __restrict__ qx,
      const _Float16* __restrict__ qW0, const _Float16* __restrict__ qW1,
      const float* __restrict__ b0, const float* __restrict__ b1,
      float* __restrict__ h0f32, float* __restrict__ h1f32,
      _Float16* __restrict__ h0f16, _Float16* __restrict__ h1f16,
      unsigned* __restrict__ bar,
      float* __restrict__ y /* [B,S,H] */, float* __restrict__ hfin /* [2,B,H] */) {
  extern __shared__ _Float16 lds[];        // A-operand staging, 64 x LDA1 halfs

  const int lane  = threadIdx.x & 31;
  const int wid   = blockIdx.x * (TPB / 32) + (threadIdx.x >> 5); // 0..127
  const int m     = wid >> 5;              // 0..3   (16 batch rows each)
  const int j     = wid & 31;              // 0..31  (16 cols of z and of n)
  const int lrow  = lane & 15;
  const int hi16  = lane >> 4;             // 0/1
  const int khalf = hi16 << 3;             // 0 / 8 halfs; unused (A and B both use kboff)
  const int kboff = hi16 << 4;             // B chunk offset (0 / 16 halfs)
  const int rbase = m * 16 + (hi16 << 3);  // C/D row base for this lane
  const int c     = j * 16 + lrow;         // output column (within H)

  const float bz0 = b0[c], bn0 = b0[Hsz + c];
  const float bz1 = b1[c], bn1 = b1[Hsz + c];

  const _Float16* Wz0 = qW0 + (size_t)c * K0;
  const _Float16* Wn0 = qW0 + (size_t)(Hsz + c) * K0;
  const _Float16* Wz1 = qW1 + (size_t)c * K1;
  const _Float16* Wn1 = qW1 + (size_t)(Hsz + c) * K1;

  for (int t = 0; t < Ssz; ++t) {
    const int p = t & 1;                   // double-buffer parity
    const _Float16* h0in   = h0f16 + p * (Bsz * Hsz);
    _Float16*       h0out  = h0f16 + (p ^ 1) * (Bsz * Hsz);
    const float*    h0inf  = h0f32 + p * (Bsz * Hsz);
    float*          h0outf = h0f32 + (p ^ 1) * (Bsz * Hsz);
    const _Float16* h1in   = h1f16 + p * (Bsz * Hsz);
    _Float16*       h1out  = h1f16 + (p ^ 1) * (Bsz * Hsz);
    const float*    h1inf  = h1f32 + p * (Bsz * Hsz);
    float*          h1outf = h1f32 + (p ^ 1) * (Bsz * Hsz);

    // ---- stage A0 = [ qx[:,t,:] | h0in ] into LDS (16B vectors) ----
    for (int u = threadIdx.x; u < Bsz * (K0 / 8); u += TPB) {
      int row = u / (K0 / 8), cc = u % (K0 / 8);
      v8h v = (cc < Isz / 8)
                  ? *(const v8h*)(qx + (size_t)row * Ssz * Isz + (size_t)t * Isz + cc * 8)
                  : *(const v8h*)(h0in + row * Hsz + (cc - Isz / 8) * 8);
      *(v8h*)(lds + row * LDA0 + cc * 8) = v;
    }
    __syncthreads();

    // ---- layer 0: gates_zn = A0 @ [Wz0;Wn0]^T, GRU update of h0 ----
    {
      v8f accz = {}, accn = {};
      const _Float16* Ar = lds + (m * 16 + lrow) * LDA0;
      for (int kk = 0; kk < K0 / 32; ++kk) {
        const int k0 = kk * 32;
        v16h a  = cat8(*(const v8h*)(Ar + k0 + kboff),
                       *(const v8h*)(Ar + k0 + kboff + 8));
        v16h bz = cat8(*(const v8h*)(Wz0 + k0 + kboff),
                       *(const v8h*)(Wz0 + k0 + kboff + 8));
        v16h bn = cat8(*(const v8h*)(Wn0 + k0 + kboff),
                       *(const v8h*)(Wn0 + k0 + kboff + 8));
        __builtin_prefetch(Wz0 + k0 + 32 + kboff, 0, 1);
        __builtin_prefetch(Wn0 + k0 + 32 + kboff, 0, 1);
        accz = __builtin_amdgcn_wmma_f32_16x16x32_f16(false, a, false, bz,
                                                      (short)0, accz, false, false);
        accn = __builtin_amdgcn_wmma_f32_16x16x32_f16(false, a, false, bn,
                                                      (short)0, accn, false, false);
      }
#pragma unroll
      for (int v = 0; v < 8; ++v) {
        const int r = rbase + v;
        float z    = 1.0f / (1.0f + __expf(-(accz[v] + bz0)));
        float n    = tanhf(accn[v] + bn0);
        float hold = h0inf[r * Hsz + c];
        float hn   = (1.0f - z) * n + z * hold;
        h0outf[r * Hsz + c] = hn;
        h0out[r * Hsz + c]  = (_Float16)hn;
        if (t == Ssz - 1) hfin[r * Hsz + c] = hn;          // h_fin layer 0
      }
    }
    grid_sync(bar, bar + 1);

    // ---- stage A1 = [ h0_new | h1in ] into LDS ----
    for (int u = threadIdx.x; u < Bsz * (K1 / 8); u += TPB) {
      int row = u / (K1 / 8), cc = u % (K1 / 8);
      v8h v = (cc < Hsz / 8)
                  ? *(const v8h*)(h0out + row * Hsz + cc * 8)
                  : *(const v8h*)(h1in + row * Hsz + (cc - Hsz / 8) * 8);
      *(v8h*)(lds + row * LDA1 + cc * 8) = v;
    }
    __syncthreads();

    // ---- layer 1: gates_zn = A1 @ [Wz1;Wn1]^T, GRU update of h1, emit y ----
    {
      v8f accz = {}, accn = {};
      const _Float16* Ar = lds + (m * 16 + lrow) * LDA1;
      for (int kk = 0; kk < K1 / 32; ++kk) {
        const int k0 = kk * 32;
        v16h a  = cat8(*(const v8h*)(Ar + k0 + kboff),
                       *(const v8h*)(Ar + k0 + kboff + 8));
        v16h bz = cat8(*(const v8h*)(Wz1 + k0 + kboff),
                       *(const v8h*)(Wz1 + k0 + kboff + 8));
        v16h bn = cat8(*(const v8h*)(Wn1 + k0 + kboff),
                       *(const v8h*)(Wn1 + k0 + kboff + 8));
        __builtin_prefetch(Wz1 + k0 + 32 + kboff, 0, 1);
        __builtin_prefetch(Wn1 + k0 + 32 + kboff, 0, 1);
        accz = __builtin_amdgcn_wmma_f32_16x16x32_f16(false, a, false, bz,
                                                      (short)0, accz, false, false);
        accn = __builtin_amdgcn_wmma_f32_16x16x32_f16(false, a, false, bn,
                                                      (short)0, accn, false, false);
      }
#pragma unroll
      for (int v = 0; v < 8; ++v) {
        const int r = rbase + v;
        float z    = 1.0f / (1.0f + __expf(-(accz[v] + bz1)));
        float n    = tanhf(accn[v] + bn1);
        float hold = h1inf[r * Hsz + c];
        float hn   = (1.0f - z) * n + z * hold;
        h1outf[r * Hsz + c] = hn;
        h1out[r * Hsz + c]  = (_Float16)hn;
        y[(size_t)r * Ssz * Hsz + (size_t)t * Hsz + c] = hn;
        if (t == Ssz - 1) hfin[Bsz * Hsz + r * Hsz + c] = hn;  // h_fin layer 1
      }
    }
    grid_sync(bar, bar + 1);
  }
}

// ---------------------------------------------------------------------------
extern "C" void kernel_launch(void* const* d_in, const int* in_sizes, int n_in,
                              void* d_out, int out_size, void* d_ws, size_t ws_size,
                              hipStream_t stream) {
  const float* x    = (const float*)d_in[0];
  const float* wih0 = (const float*)d_in[1];
  const float* whh0 = (const float*)d_in[2];
  const float* bih0 = (const float*)d_in[3];
  const float* bhh0 = (const float*)d_in[4];
  const float* wih1 = (const float*)d_in[5];
  const float* whh1 = (const float*)d_in[6];
  const float* bih1 = (const float*)d_in[7];
  const float* bhh1 = (const float*)d_in[8];

  char*  ws  = (char*)d_ws;
  size_t off = 0;
  auto carve = [&](size_t bytes) -> char* {
    char* p = ws + off;
    off = (off + bytes + 255) & ~(size_t)255;
    return p;
  };
  _Float16* qx    = (_Float16*)carve((size_t)Bsz * Ssz * Isz * 2);  // 32 MB
  _Float16* qW0   = (_Float16*)carve((size_t)1024 * K0 * 2);
  _Float16* qW1   = (_Float16*)carve((size_t)1024 * K1 * 2);
  float*    b0    = (float*)carve(1024 * 4);
  float*    b1    = (float*)carve(1024 * 4);
  float*    h0f32 = (float*)carve((size_t)2 * Bsz * Hsz * 4);
  float*    h1f32 = (float*)carve((size_t)2 * Bsz * Hsz * 4);
  _Float16* h0f16 = (_Float16*)carve((size_t)2 * Bsz * Hsz * 2);
  _Float16* h1f16 = (_Float16*)carve((size_t)2 * Bsz * Hsz * 2);
  unsigned* bar   = (unsigned*)carve(256);

  float* y    = (float*)d_out;
  float* hfin = y + (size_t)Bsz * Ssz * Hsz;

  const int nx = Bsz * Ssz * Isz;
  k_quant_x<<<nx / (4 * 256), 256, 0, stream>>>(x, qx, nx);
  k_pack<<<2048, 256, 0, stream>>>(wih0, whh0, bih0, bhh0,
                                   wih1, whh1, bih1, bhh1, qW0, qW1, b0, b1);
  k_init<<<(2 * Bsz * Hsz + 255) / 256, 256, 0, stream>>>(h0f32, h1f32, h0f16, h1f16, bar);

  const size_t ldsBytes = (size_t)Bsz * LDA1 * sizeof(_Float16);   // 132,096 B
  k_gru<<<NWG, TPB, ldsBytes, stream>>>(qx, qW0, qW1, b0, b1,
                                        h0f32, h1f32, h0f16, h1f16, bar, y, hfin);
}
